// CoarseMatching_70987219468985
// MI455X (gfx1250) — matrix operand order repaired and checked by reading, not compiled, not run on hardware
//
#include <hip/hip_runtime.h>
#include <hip/hip_bf16.h>

typedef __attribute__((ext_vector_type(16))) __bf16 v16bf;
typedef __attribute__((ext_vector_type(8)))  float  v8f;

#define N_ROWS 8192
#define DDIM   256
#define TILE   128
#define KT     32
#define PADK   40      // padded K stride (elements) -> 80B rows, conflict-free b128 LDS reads
#define NBINS  8192
#define CAP    8192

struct Frag { unsigned u[8]; };   // 16 bf16 packed as 8 dwords

struct GemmLds {
  __align__(16) unsigned short Ah[TILE * PADK];
  __align__(16) unsigned short Al[TILE * PADK];
  __align__(16) unsigned short Bh[TILE * PADK];
  __align__(16) unsigned short Bl[TILE * PADK];
};

static __device__ __forceinline__ v8f wmma_bf16(const Frag& a, const Frag& b, v8f c) {
  union U { Frag f; v16bf v; };
  U ua, ub; ua.f = a; ub.f = b;
  return __builtin_amdgcn_wmma_f32_16x16x32_bf16(
      /*neg_a=*/false, ua.v, /*neg_b=*/false, ub.v,
      /*c_mod=*/(short)0, c, /*reuse_a=*/false, /*reuse_b=*/false);
}

// Split f32 into bf16 hi (truncate) + bf16 lo (residual), store 4 at a time packed.
static __device__ __forceinline__ void cvt_store4(float4 v, unsigned short* ph, unsigned short* pl) {
  unsigned u0 = __float_as_uint(v.x), u1 = __float_as_uint(v.y);
  unsigned u2 = __float_as_uint(v.z), u3 = __float_as_uint(v.w);
  float r0 = v.x - __uint_as_float(u0 & 0xFFFF0000u);
  float r1 = v.y - __uint_as_float(u1 & 0xFFFF0000u);
  float r2 = v.z - __uint_as_float(u2 & 0xFFFF0000u);
  float r3 = v.w - __uint_as_float(u3 & 0xFFFF0000u);
  unsigned* H = (unsigned*)ph;
  unsigned* L = (unsigned*)pl;
  H[0] = (u1 & 0xFFFF0000u) | (u0 >> 16);
  H[1] = (u3 & 0xFFFF0000u) | (u2 >> 16);
  unsigned s0 = __float_as_uint(r0), s1 = __float_as_uint(r1);
  unsigned s2 = __float_as_uint(r2), s3 = __float_as_uint(r3);
  L[0] = (s1 & 0xFFFF0000u) | (s0 >> 16);
  L[1] = (s3 & 0xFFFF0000u) | (s2 >> 16);
}

// A fragment 16x32 bf16: lane<16 -> K 0..7 & 16..23; lane>=16 -> K 8..15 & 24..31 (row = rbase + lane&15)
static __device__ __forceinline__ Frag load_a_frag(const unsigned short* base, int rbase, int lane) {
  int r  = rbase + (lane & 15);
  int k0 = (lane < 16) ? 0 : 8;
  uint4 a = *(const uint4*)(base + r * PADK + k0);
  uint4 b = *(const uint4*)(base + r * PADK + k0 + 16);
  Frag f;
  f.u[0] = a.x; f.u[1] = a.y; f.u[2] = a.z; f.u[3] = a.w;
  f.u[4] = b.x; f.u[5] = b.y; f.u[6] = b.z; f.u[7] = b.w;
  return f;
}

// B fragment 32x16 bf16 (B row n holds K contiguous): lane<16 -> K 0..7 & 16..23; lane>=16 -> K 8..15 & 24..31
static __device__ __forceinline__ Frag load_b_frag(const unsigned short* base, int nbase, int lane) {
  int n  = nbase + (lane & 15);
  int k0 = (lane < 16) ? 0 : 8;
  uint4 a = *(const uint4*)(base + n * PADK + k0);
  uint4 b = *(const uint4*)(base + n * PADK + k0 + 16);
  Frag f;
  f.u[0] = a.x; f.u[1] = a.y; f.u[2] = a.z; f.u[3] = a.w;
  f.u[4] = b.x; f.u[5] = b.y; f.u[6] = b.z; f.u[7] = b.w;
  return f;
}

// 128x128 output tile: 8 waves, wave w computes rows [16w,16w+16) x 128 cols (8 WMMA subtiles).
// acc[nb][g]: row = wave*16 + g + 8*(lane>>4), col = nb*16 + (lane&15). bf16x3 compensated product.
static __device__ __forceinline__ void gemm_tile(
    const float* __restrict__ ref, const float* __restrict__ src,
    int tm, int tn, GemmLds& L, v8f (&acc)[8])
{
  const int tid  = threadIdx.x;
  const int lane = tid & 31;
  const int wave = tid >> 5;
  const int rbase = wave * 16;
#pragma unroll
  for (int nb = 0; nb < 8; ++nb)
#pragma unroll
    for (int g = 0; g < 8; ++g) acc[nb][g] = 0.0f;

  const int row = tid >> 1;            // 2 threads per row, 16 floats each
  const int cb  = (tid & 1) * 16;
  const float* gA = ref + (size_t)(tm * TILE + row) * DDIM + cb;
  const float* gB = src + (size_t)(tn * TILE + row) * DDIM + cb;

  for (int kk = 0; kk < DDIM; kk += KT) {
    __syncthreads();
#pragma unroll
    for (int i = 0; i < 4; ++i) {
      float4 va = *(const float4*)(gA + kk + 4 * i);
      float4 vb = *(const float4*)(gB + kk + 4 * i);
      cvt_store4(va, &L.Ah[row * PADK + cb + 4 * i], &L.Al[row * PADK + cb + 4 * i]);
      cvt_store4(vb, &L.Bh[row * PADK + cb + 4 * i], &L.Bl[row * PADK + cb + 4 * i]);
    }
    __syncthreads();
    Frag ah = load_a_frag(L.Ah, rbase, lane);
    Frag al = load_a_frag(L.Al, rbase, lane);
#pragma unroll
    for (int nb = 0; nb < 8; ++nb) {
      Frag bh = load_b_frag(L.Bh, nb * 16, lane);
      Frag bl = load_b_frag(L.Bl, nb * 16, lane);
      acc[nb] = wmma_bf16(ah, bh, acc[nb]);
      acc[nb] = wmma_bf16(ah, bl, acc[nb]);
      acc[nb] = wmma_bf16(al, bh, acc[nb]);
    }
  }
  __syncthreads();
}

// ---------------- kernels ----------------

__global__ __launch_bounds__(256) void k_zero(unsigned* hist, unsigned* ctrs) {
  int i = blockIdx.x * 256 + threadIdx.x;
  if (i < NBINS) hist[i] = 0u;
  if (i < 8)     ctrs[i] = 0u;
}

__global__ __launch_bounds__(256) void k_norms(const float* __restrict__ ref,
                                               const float* __restrict__ src,
                                               float* __restrict__ rn, float* __restrict__ sn) {
  int wave = (int)((blockIdx.x * blockDim.x + threadIdx.x) >> 5);
  int lane = threadIdx.x & 31;
  if (wave >= 2 * N_ROWS) return;
  const float* base = (wave < N_ROWS) ? (ref + (size_t)wave * DDIM)
                                      : (src + (size_t)(wave - N_ROWS) * DDIM);
  const float4* p = (const float4*)base;
  float s = 0.f;
#pragma unroll
  for (int i = 0; i < 2; ++i) {
    float4 v = p[lane + 32 * i];
    s += v.x * v.x + v.y * v.y + v.z * v.z + v.w * v.w;
  }
  for (int o = 16; o > 0; o >>= 1) s += __shfl_xor(s, o, 32);
  if (lane == 0) { if (wave < N_ROWS) rn[wave] = s; else sn[wave - N_ROWS] = s; }
}

__global__ __launch_bounds__(256) void k_sums(
    const float* __restrict__ ref, const float* __restrict__ src,
    const float* __restrict__ rn, const float* __restrict__ sn,
    float* __restrict__ rowpart, float* __restrict__ colpart)
{
  __shared__ GemmLds L;
  __shared__ float s_rn[TILE], s_sn[TILE];
  __shared__ float wcol[8 * TILE];
  __shared__ float lrow[TILE];
  const int tm = blockIdx.x, tn = blockIdx.y;
  const int tid = threadIdx.x, lane = tid & 31, wave = tid >> 5;
  if (tid < TILE) { s_rn[tid] = rn[tm * TILE + tid]; s_sn[tid] = sn[tn * TILE + tid]; }
  v8f acc[8];
  gemm_tile(ref, src, tm, tn, L, acc);
  const int rbase = wave * 16, hb = lane >> 4, cl = lane & 15;
#pragma unroll
  for (int nb = 0; nb < 8; ++nb)
#pragma unroll
    for (int g = 0; g < 8; ++g) {
      int r = rbase + g + 8 * hb, c = nb * 16 + cl;
      float d2 = s_rn[r] + s_sn[c] - 2.0f * acc[nb][g];
      acc[nb][g] = expf(-d2);
    }
  // deterministic row partials (reduce over 16 lanes within half)
#pragma unroll
  for (int g = 0; g < 8; ++g) {
    float s = 0.f;
#pragma unroll
    for (int nb = 0; nb < 8; ++nb) s += acc[nb][g];
    s += __shfl_xor(s, 1, 32); s += __shfl_xor(s, 2, 32);
    s += __shfl_xor(s, 4, 32); s += __shfl_xor(s, 8, 32);
    if (cl == 0) lrow[rbase + g + 8 * hb] = s;
  }
  // deterministic col partials (reduce over halves, then over waves via LDS)
#pragma unroll
  for (int nb = 0; nb < 8; ++nb) {
    float s = 0.f;
#pragma unroll
    for (int g = 0; g < 8; ++g) s += acc[nb][g];
    s += __shfl_xor(s, 16, 32);
    if (hb == 0) wcol[wave * TILE + nb * 16 + cl] = s;
  }
  __syncthreads();
  if (tid < TILE) {
    float s = 0.f;
#pragma unroll
    for (int w = 0; w < 8; ++w) s += wcol[w * TILE + tid];
    colpart[(size_t)tm * N_ROWS + tn * TILE + tid] = s;
    rowpart[(size_t)tn * N_ROWS + tm * TILE + tid] = lrow[tid];
  }
}

__global__ __launch_bounds__(256) void k_reduce(const float* __restrict__ rowpart,
                                                const float* __restrict__ colpart,
                                                float* __restrict__ rowsum,
                                                float* __restrict__ colsum) {
  int i = blockIdx.x * 256 + threadIdx.x;   // 8192 threads total
  float s = 0.f, c = 0.f;
  for (int t = 0; t < 64; ++t) {
    s += rowpart[(size_t)t * N_ROWS + i];
    c += colpart[(size_t)t * N_ROWS + i];
  }
  rowsum[i] = s; colsum[i] = c;
}

__global__ __launch_bounds__(256) void k_hist(
    const float* __restrict__ ref, const float* __restrict__ src,
    const float* __restrict__ rn, const float* __restrict__ sn,
    const float* __restrict__ rowsum, const float* __restrict__ colsum,
    unsigned* __restrict__ hist)
{
  __shared__ GemmLds L;
  __shared__ float s_rn[TILE], s_sn[TILE], s_R[TILE], s_C[TILE];
  __shared__ unsigned hl[NBINS];
  const int tm = blockIdx.x, tn = blockIdx.y;
  const int tid = threadIdx.x, lane = tid & 31, wave = tid >> 5;
  for (int i = tid; i < NBINS; i += 256) hl[i] = 0u;
  if (tid < TILE) {
    s_rn[tid] = rn[tm * TILE + tid]; s_sn[tid] = sn[tn * TILE + tid];
    s_R[tid] = rowsum[tm * TILE + tid]; s_C[tid] = colsum[tn * TILE + tid];
  }
  v8f acc[8];
  gemm_tile(ref, src, tm, tn, L, acc);
  const int rbase = wave * 16, hb = lane >> 4, cl = lane & 15;
#pragma unroll
  for (int nb = 0; nb < 8; ++nb)
#pragma unroll
    for (int g = 0; g < 8; ++g) {
      int r = rbase + g + 8 * hb, c = nb * 16 + cl;
      float e = expf(-(s_rn[r] + s_sn[c] - 2.0f * acc[nb][g]));
      float f = e * e / ((s_R[r] + 1e-8f) * (s_C[c] + 1e-8f));
      unsigned key = __float_as_uint(f) >> 19;  // positive floats sort by bit pattern
      if (key > NBINS - 1u) key = NBINS - 1u;
      atomicAdd(&hl[key], 1u);
    }
  __syncthreads();
  for (int i = tid; i < NBINS; i += 256) { unsigned v = hl[i]; if (v) atomicAdd(&hist[i], v); }
}

__global__ __launch_bounds__(256) void k_thresh(const unsigned* __restrict__ hist,
                                                unsigned* __restrict__ ctrs,
                                                const int* __restrict__ Kp) {
  __shared__ unsigned chunk[256];
  int tid = threadIdx.x;
  unsigned s = 0;
  for (int i = 0; i < 32; ++i) s += hist[tid * 32 + i];
  chunk[tid] = s;
  __syncthreads();
  if (tid == 0) {
    unsigned K = (unsigned)(*Kp);
    unsigned cum = 0, bin = 0; bool found = false;
    for (int c = 255; c >= 0 && !found; --c) {
      if (cum + chunk[c] >= K) {
        for (int b = 31; b >= 0; --b) {
          unsigned h = hist[c * 32 + b];
          if (cum + h >= K) { bin = (unsigned)(c * 32 + b); found = true; break; }
          cum += h;
        }
      } else cum += chunk[c];
    }
    ctrs[0] = bin;   // inclusive threshold bin
    ctrs[1] = 0u;    // candidate counter
  }
}

__global__ __launch_bounds__(256) void k_collect(
    const float* __restrict__ ref, const float* __restrict__ src,
    const float* __restrict__ rn, const float* __restrict__ sn,
    const float* __restrict__ rowsum, const float* __restrict__ colsum,
    unsigned* __restrict__ ctrs, float* __restrict__ cscore, unsigned* __restrict__ cidx)
{
  __shared__ GemmLds L;
  __shared__ float s_rn[TILE], s_sn[TILE], s_R[TILE], s_C[TILE];
  const int tm = blockIdx.x, tn = blockIdx.y;
  const int tid = threadIdx.x, lane = tid & 31, wave = tid >> 5;
  if (tid < TILE) {
    s_rn[tid] = rn[tm * TILE + tid]; s_sn[tid] = sn[tn * TILE + tid];
    s_R[tid] = rowsum[tm * TILE + tid]; s_C[tid] = colsum[tn * TILE + tid];
  }
  const unsigned tb = ctrs[0];
  v8f acc[8];
  gemm_tile(ref, src, tm, tn, L, acc);
  const int rbase = wave * 16, hb = lane >> 4, cl = lane & 15;
#pragma unroll
  for (int nb = 0; nb < 8; ++nb)
#pragma unroll
    for (int g = 0; g < 8; ++g) {
      int r = rbase + g + 8 * hb, c = nb * 16 + cl;
      float e = expf(-(s_rn[r] + s_sn[c] - 2.0f * acc[nb][g]));
      float f = e * e / ((s_R[r] + 1e-8f) * (s_C[c] + 1e-8f));
      unsigned key = __float_as_uint(f) >> 19;
      if (key >= tb) {
        unsigned slot = atomicAdd(&ctrs[1], 1u);
        if (slot < CAP) {
          cscore[slot] = f;
          cidx[slot] = ((unsigned)(tm * TILE + r) << 13) | (unsigned)(tn * TILE + c);
        }
      }
    }
}

__global__ __launch_bounds__(256) void k_topk(
    const float* __restrict__ cscore, const unsigned* __restrict__ cidx,
    const unsigned* __restrict__ ctrs, const int* __restrict__ Kp,
    int* __restrict__ oRef, int* __restrict__ oSrc, float* __restrict__ oSc)
{
  __shared__ float sc[CAP];
  __shared__ unsigned si[CAP];
  __shared__ float rs[256];
  __shared__ int rp[256];
  const int tid = threadIdx.x;
  unsigned n = ctrs[1]; if (n > CAP) n = CAP;
  for (unsigned i = tid; i < CAP; i += 256) {
    if (i < n) { sc[i] = cscore[i]; si[i] = cidx[i]; }
    else       { sc[i] = -1e30f;    si[i] = 0xFFFFFFFFu; }
  }
  __syncthreads();
  const int K = *Kp;
  for (int it = 0; it < K; ++it) {
    float best = sc[tid]; int bpos = tid;
    for (int i = tid + 256; i < CAP; i += 256) {
      float s = sc[i];
      if (s > best || (s == best && si[i] < si[bpos])) { best = s; bpos = i; }
    }
    rs[tid] = best; rp[tid] = bpos;
    __syncthreads();
    for (int o = 128; o > 0; o >>= 1) {
      if (tid < o) {
        float s2 = rs[tid + o]; int p2 = rp[tid + o];
        if (s2 > rs[tid] || (s2 == rs[tid] && si[p2] < si[rp[tid]])) { rs[tid] = s2; rp[tid] = p2; }
      }
      __syncthreads();
    }
    if (tid == 0) {
      int pos = rp[0]; unsigned fi = si[pos];
      oRef[it] = (int)(fi >> 13);
      oSrc[it] = (int)(fi & 8191u);
      oSc[it]  = rs[0];
      sc[pos] = -2e30f;  // consume
    }
    __syncthreads();
  }
}

extern "C" void kernel_launch(void* const* d_in, const int* in_sizes, int n_in,
                              void* d_out, int out_size, void* d_ws, size_t ws_size,
                              hipStream_t stream)
{
  const float* ref = (const float*)d_in[0];
  const float* src = (const float*)d_in[1];
  const int*   Kp  = (const int*)d_in[2];

  float* w       = (float*)d_ws;
  float* rowsum  = w;                       // 8192
  float* colsum  = w + 8192;                // 8192
  float* rn      = w + 16384;               // 8192
  float* sn      = w + 24576;               // 8192
  unsigned* hist = (unsigned*)(w + 32768);  // 8192
  unsigned* ctrs = (unsigned*)(w + 40960);  // 8
  float* rowpart = w + 40968;               // 64*8192
  float* colpart = rowpart + 64 * N_ROWS;   // 64*8192
  float* cscore  = colpart + 64 * N_ROWS;   // CAP
  unsigned* cidx = (unsigned*)(cscore + CAP);

  int*   oRef = (int*)d_out;
  int*   oSrc = oRef + 256;
  float* oSc  = (float*)d_out + 512;

  dim3 grid(64, 64);
  k_zero   <<<32,   256, 0, stream>>>(hist, ctrs);
  k_norms  <<<2048, 256, 0, stream>>>(ref, src, rn, sn);
  k_sums   <<<grid, 256, 0, stream>>>(ref, src, rn, sn, rowpart, colpart);
  k_reduce <<<32,   256, 0, stream>>>(rowpart, colpart, rowsum, colsum);
  k_hist   <<<grid, 256, 0, stream>>>(ref, src, rn, sn, rowsum, colsum, hist);
  k_thresh <<<1,    256, 0, stream>>>(hist, ctrs, Kp);
  k_collect<<<grid, 256, 0, stream>>>(ref, src, rn, sn, rowsum, colsum, ctrs, cscore, cidx);
  k_topk   <<<1,    256, 0, stream>>>(cscore, cidx, ctrs, Kp, oRef, oSrc, oSc);
}
